// TransformerBlock_51453708206312
// MI455X (gfx1250) — matrix operand order repaired and checked by reading, not compiled, not run on hardware
//
#include <hip/hip_runtime.h>
#include <hip/hip_bf16.h>
#include <math.h>

// Problem dims (fixed by the reference)
#define DM   1024          // d_model
#define NH   16            // heads
#define HD   64            // head dim
#define DFF  4096          // ffn dim
#define BB   2             // batch
#define TT   2048          // seq len
#define BT   (BB*TT)       // 4096 rows

typedef __bf16 bf16;
typedef __attribute__((ext_vector_type(16))) __bf16 v16bf;
typedef __attribute__((ext_vector_type(8)))  float  v8f;

__device__ __forceinline__ v8f wmma_bf16(v16bf a, v16bf b, v8f c) {
  // D = A(16x32 bf16) * B(32x16 bf16) + C(16x16 f32)
  return __builtin_amdgcn_wmma_f32_16x16x32_bf16(false, a, false, b, (short)0, c,
                                                 false, false);
}

// ---------------------------------------------------------------------------
// gfx1250 async global->LDS copy path (ASYNCcnt-tracked, bypasses VGPRs).
// Probe (round-3 diagnostic) showed the builtin exists with signature:
//   (int __vector_size(16) AS1*, <lds ptr>, imm offset, imm cpol)
// Guarded so a toolchain without the builtin still compiles via uint4 staging.
// ---------------------------------------------------------------------------
#if defined(__has_builtin)
#if __has_builtin(__builtin_amdgcn_global_load_async_to_lds_b128) && \
    __has_builtin(__builtin_amdgcn_s_wait_asynccnt)
#define USE_ASYNC_COPY 1
#endif
#endif
#ifndef USE_ASYNC_COPY
#define USE_ASYNC_COPY 0
#endif

#if USE_ASYNC_COPY
typedef int v4i_b128 __attribute__((vector_size(16)));   // GCC-style, per diag
typedef __attribute__((address_space(1))) v4i_b128 gv4_t;  // global (AS1)
typedef __attribute__((address_space(3))) v4i_b128 lv4_t;  // LDS (AS3)
__device__ __forceinline__ void async_copy16(const void* g, void* l) {
  __builtin_amdgcn_global_load_async_to_lds_b128((gv4_t*)g, (lv4_t*)l, 0, 0);
}
#endif

// ---------------------------------------------------------------------------
// f32 [K][N] -> bf16 transposed [N][K]  (weights staged once per launch).
// LDS 32x32 tile transpose keeps both the read and the write coalesced.
// ---------------------------------------------------------------------------
__global__ __launch_bounds__(256)
void cvt_transpose_kernel(const float* __restrict__ w, bf16* __restrict__ wt,
                          int K, int N) {
  __shared__ bf16 tile[32][33];
  const int k0 = blockIdx.y * 32;
  const int n0 = blockIdx.x * 32;
  const int tx = threadIdx.x & 31;
  const int ty = threadIdx.x >> 5;    // 8 row-groups
#pragma unroll
  for (int i = 0; i < 32; i += 8)
    tile[ty + i][tx] = (bf16)w[(size_t)(k0 + ty + i) * N + n0 + tx];
  __syncthreads();
#pragma unroll
  for (int i = 0; i < 32; i += 8)
    wt[(size_t)(n0 + ty + i) * K + k0 + tx] = tile[tx][ty + i];
}

// ---------------------------------------------------------------------------
// LayerNorm: one block per row of 1024, outputs bf16
// ---------------------------------------------------------------------------
__global__ __launch_bounds__(256)
void layernorm_kernel(const float* __restrict__ x, const float* __restrict__ g,
                      const float* __restrict__ b, bf16* __restrict__ out) {
  __shared__ float s1[8], s2[8];
  const int row = blockIdx.x;
  const float* xr = x + (size_t)row * DM;
  const int t = threadIdx.x;
  float v[4], a1 = 0.f, a2 = 0.f;
#pragma unroll
  for (int i = 0; i < 4; ++i) {
    float f = xr[t + 256 * i];
    v[i] = f; a1 += f; a2 += f * f;
  }
#pragma unroll
  for (int off = 16; off; off >>= 1) {
    a1 += __shfl_down(a1, off, 32);
    a2 += __shfl_down(a2, off, 32);
  }
  if ((t & 31) == 0) { s1[t >> 5] = a1; s2[t >> 5] = a2; }
  __syncthreads();
  if (t == 0) {
    float t1 = 0.f, t2 = 0.f;
#pragma unroll
    for (int i = 0; i < 8; ++i) { t1 += s1[i]; t2 += s2[i]; }
    s1[0] = t1; s2[0] = t2;
  }
  __syncthreads();
  const float mu  = s1[0] * (1.0f / DM);
  const float var = s2[0] * (1.0f / DM) - mu * mu;
  const float rstd = rsqrtf(var + 1e-5f);
#pragma unroll
  for (int i = 0; i < 4; ++i) {
    int idx = t + 256 * i;
    out[(size_t)row * DM + idx] = (bf16)((v[i] - mu) * rstd * g[idx] + b[idx]);
  }
}

// ---------------------------------------------------------------------------
// bf16 WMMA GEMM: C[M,N] = A[M,K] @ Bt[N,K]^T (+ fused epilogue)
// B is supplied TRANSPOSED ([N][K]) so every WMMA B-fragment is a per-lane
// contiguous 32B read (2x ds_load_b128) instead of 16 strided u16 gathers.
// Block = 128 threads (4 waves). Block tile 64x64, K step 32.
// Epilogues: 0 = bias + split to q/k [B,H,T,64] and v [B,H,64,T] bf16
//            1 = bias + residual(f32)      -> f32
//            2 = bias + exact GELU         -> bf16
//            3 = bias + residual(f32)      -> f32 (final output)
// ---------------------------------------------------------------------------
template <int EPI>
__global__ __launch_bounds__(128)
void gemm_bf16_kernel(const bf16* __restrict__ A, const bf16* __restrict__ Bt,
                      const float* __restrict__ bias, const float* __restrict__ resid,
                      float* __restrict__ outF,
                      bf16* __restrict__ o0, bf16* __restrict__ o1, bf16* __restrict__ o2,
                      int M, int N, int K) {
  __shared__ __align__(16) bf16 As[64][40];  // 64 M-rows x 32 K, padded (80B rows)
  __shared__ __align__(16) bf16 Bs[64][40];  // 64 N-rows x 32 K, padded

  const int tid   = threadIdx.x;
  const int wave  = tid >> 5;
  const int lane  = tid & 31;
  const int col16 = lane & 15;
  const int hlf   = lane >> 4;       // 0: lanes 0-15, 1: lanes 16-31
  const int m0 = blockIdx.y * 64;
  const int n0 = blockIdx.x * 64;
  const int sr = tid >> 1, sc = (tid & 1) * 16;   // staging: 2 threads/row, 32B each

  v8f acc[4];
#pragma unroll
  for (int j = 0; j < 4; ++j)
#pragma unroll
    for (int r = 0; r < 8; ++r) acc[j][r] = 0.f;

  const bf16* aSrc = A  + (size_t)(m0 + sr) * K + sc;
  const bf16* bSrc = Bt + (size_t)(n0 + sr) * K + sc;

  for (int k0 = 0; k0 < K; k0 += 32) {
#if USE_ASYNC_COPY
    // ASYNCcnt-tracked copies straight into LDS (no VGPR round-trip).
    async_copy16(aSrc + k0,     &As[sr][sc]);
    async_copy16(aSrc + k0 + 8, &As[sr][sc + 8]);
    async_copy16(bSrc + k0,     &Bs[sr][sc]);
    async_copy16(bSrc + k0 + 8, &Bs[sr][sc + 8]);
#else
    { // stage A tile 64x32
      const uint4* src = (const uint4*)(aSrc + k0);
      uint4* dst = (uint4*)(&As[sr][sc]);
      dst[0] = src[0]; dst[1] = src[1];
    }
    { // stage B^T tile 64x32 (identical pattern)
      const uint4* src = (const uint4*)(bSrc + k0);
      uint4* dst = (uint4*)(&Bs[sr][sc]);
      dst[0] = src[0]; dst[1] = src[1];
    }
#endif
    if (k0 + 32 < K) {  // prefetch next tiles into cache (global_prefetch_b8)
      __builtin_prefetch(aSrc + k0 + 32, 0, 1);
      __builtin_prefetch(bSrc + k0 + 32, 0, 1);
    }
#if USE_ASYNC_COPY
    __builtin_amdgcn_s_wait_asynccnt(0);
#endif
    __syncthreads();

    // A fragment (ISA 7.12.2): row = lane%16, K set depends on lane half
    v16bf af;
    {
      const bf16* ap = &As[wave * 16 + col16][0];
      const int kb = hlf * 16;
#pragma unroll
      for (int i = 0; i < 8; ++i) { af[i] = ap[kb + i]; af[8 + i] = ap[kb + 8 + i]; }
    }
    // B fragments: lane = column n, K 0-15 (lanes 0-15) / 16-31 (lanes 16-31),
    // contiguous per lane in Bs -> 2x ds_load_b128.
#pragma unroll
    for (int j = 0; j < 4; ++j) {
      v16bf bfr;
      const bf16* bp = &Bs[j * 16 + col16][hlf * 16];
#pragma unroll
      for (int i = 0; i < 16; ++i) bfr[i] = bp[i];
      acc[j] = wmma_bf16(af, bfr, acc[j]);
    }
    __syncthreads();
  }

  // Epilogue. C-frag: lane holds col = col16, rows hlf*8 + r.
  const int rbase = m0 + wave * 16 + hlf * 8;
#pragma unroll
  for (int j = 0; j < 4; ++j) {
    const int col = n0 + j * 16 + col16;
#pragma unroll
    for (int r = 0; r < 8; ++r) {
      const int row = rbase + r;
      float val = acc[j][r] + bias[col];
      if (EPI == 0) {             // qkv split: col in [0,3072)
        const int which = col >> 10;
        const int rem = col & 1023;
        const int h = rem >> 6, d = rem & 63;
        const int bb = row >> 11, t = row & 2047;
        if (which == 2) {         // V stored [B,H,HD,T] (time-major last)
          o2[(((size_t)bb * NH + h) * HD + d) * TT + t] = (bf16)val;
        } else {
          bf16* dst = (which == 0) ? o0 : o1;
          dst[(((size_t)bb * NH + h) * TT + t) * HD + d] = (bf16)val;
        }
      } else if (EPI == 1 || EPI == 3) {
        outF[(size_t)row * N + col] = val + resid[(size_t)row * N + col];
      } else {                    // EPI == 2: exact GELU
        float gv = 0.5f * val * (1.0f + erff(val * 0.70710678118654752f));
        o0[(size_t)row * N + col] = (bf16)gv;
      }
    }
  }
}

// ---------------------------------------------------------------------------
// Flash attention: one wave per 16-query tile.
// Q,K are [B,H,T,64] bf16; V is [B,H,64,T] bf16 (transposed) so that P@V
// B-fragments are per-lane contiguous 32B reads (global_load_b128).
// Streams keys 32 at a time: 4 WMMA for S = Q K^T, online softmax, then
// P (via LDS transpose to A-layout) @ V with 4 WMMA. Output [B,T,DM] bf16.
// ---------------------------------------------------------------------------
__global__ __launch_bounds__(128)
void attention_kernel(const bf16* __restrict__ Q, const bf16* __restrict__ Kt,
                      const bf16* __restrict__ V, bf16* __restrict__ O) {
  __shared__ __align__(16) bf16 Pbuf[4][16][32];  // per-wave P staging

  const int wave = threadIdx.x >> 5;
  const int lane = threadIdx.x & 31;
  const int col16 = lane & 15;
  const int hlf = lane >> 4;
  const int tile = blockIdx.x * 4 + wave;       // 4096 tiles total
  const int qt = tile & 127;                    // 128 q-tiles of 16
  const int bh = tile >> 7;                     // 0..31 (b*16+h)
  const size_t base = (size_t)bh * TT * HD;
  const bf16* Qp = Q + base;
  const bf16* Kp = Kt + base;
  const bf16* Vp = V + base;                    // [64][TT] within this (b,h)
  const int q0 = qt * 16;

  // Q fragments for K-dim 0..31 and 32..63, held in registers for whole tile.
  v16bf qf[2];
  {
    const bf16* qr = Qp + (size_t)(q0 + col16) * HD;
#pragma unroll
    for (int f = 0; f < 2; ++f) {
      const int kb = f * 32 + hlf * 16;
#pragma unroll
      for (int i = 0; i < 8; ++i) { qf[f][i] = qr[kb + i]; qf[f][8 + i] = qr[kb + 8 + i]; }
    }
  }

  v8f o[4];
  float m_i[8], l_i[8];
#pragma unroll
  for (int j = 0; j < 4; ++j)
#pragma unroll
    for (int r = 0; r < 8; ++r) o[j][r] = 0.f;
#pragma unroll
  for (int r = 0; r < 8; ++r) { m_i[r] = -__builtin_inff(); l_i[r] = 0.f; }

  const float scale = 0.125f;  // 1/sqrt(64)
  const int kend = q0 + 16;

  for (int kb0 = 0; kb0 < kend; kb0 += 32) {
    // ---- S = Q K^T for 32 keys (two 16x16 C fragments) ----
    v8f s[2];
#pragma unroll
    for (int sc = 0; sc < 2; ++sc) {
#pragma unroll
      for (int r = 0; r < 8; ++r) s[sc][r] = 0.f;
      const bf16* kr = Kp + (size_t)(kb0 + sc * 16 + col16) * HD;  // key row
#pragma unroll
      for (int f = 0; f < 2; ++f) {
        v16bf bfr;                       // B[kdim][key]: contiguous in key row
        const int kb = f * 32 + hlf * 16;
#pragma unroll
        for (int i = 0; i < 16; ++i) bfr[i] = kr[kb + i];
        s[sc] = wmma_bf16(qf[f], bfr, s[sc]);
      }
    }

    // ---- causal mask + scale + online softmax (branch-free: EXEC stays 1s) ----
#pragma unroll
    for (int r = 0; r < 8; ++r) {
      const int qrow = q0 + hlf * 8 + r;
      float mr = m_i[r];
#pragma unroll
      for (int sc = 0; sc < 2; ++sc) {
        const int key = kb0 + sc * 16 + col16;
        float sv = s[sc][r] * scale;
        sv = (key > qrow) ? -__builtin_inff() : sv;
        s[sc][r] = sv;
        mr = fmaxf(mr, sv);
      }
#pragma unroll
      for (int off = 1; off < 16; off <<= 1) mr = fmaxf(mr, __shfl_xor(mr, off, 32));
      const float corr = __expf(m_i[r] - mr);
      float ps = 0.f;
#pragma unroll
      for (int sc = 0; sc < 2; ++sc) {
        float pv = __expf(s[sc][r] - mr);
        s[sc][r] = pv;
        ps += pv;
      }
#pragma unroll
      for (int off = 1; off < 16; off <<= 1) ps += __shfl_xor(ps, off, 32);
      l_i[r] = l_i[r] * corr + ps;
      m_i[r] = mr;
#pragma unroll
      for (int j = 0; j < 4; ++j) o[j][r] *= corr;
    }

    // ---- transpose P: C-layout -> A-layout through per-wave LDS ----
    bf16 (*P)[32] = Pbuf[wave];
#pragma unroll
    for (int sc = 0; sc < 2; ++sc)
#pragma unroll
      for (int r = 0; r < 8; ++r)
        P[hlf * 8 + r][sc * 16 + col16] = (bf16)s[sc][r];
    asm volatile("s_wait_dscnt 0x0" ::: "memory");  // DS stores visible to this wave

    v16bf pf;
    {
      const bf16* pr = &P[col16][0];
      const int kb = hlf * 16;
#pragma unroll
      for (int i = 0; i < 8; ++i) { pf[i] = pr[kb + i]; pf[8 + i] = pr[kb + 8 + i]; }
    }

    // ---- O += P @ V : V^T rows are per-lane contiguous -> global_load_b128 ----
#pragma unroll
    for (int j = 0; j < 4; ++j) {
      v16bf vf;
      const bf16* vp = Vp + (size_t)(j * 16 + col16) * TT + kb0 + hlf * 16;
#pragma unroll
      for (int i = 0; i < 16; ++i) vf[i] = vp[i];
      o[j] = wmma_bf16(pf, vf, o[j]);
    }
  }

  // ---- normalize and write out as [B,T,DM] bf16 ----
  const int h = bh & 15, bb = bh >> 4;
#pragma unroll
  for (int r = 0; r < 8; ++r) {
    const float inv = 1.0f / l_i[r];
#pragma unroll
    for (int j = 0; j < 4; ++j) o[j][r] *= inv;
  }
#pragma unroll
  for (int j = 0; j < 4; ++j) {
    const int d = j * 16 + col16;
#pragma unroll
    for (int r = 0; r < 8; ++r) {
      const int t = q0 + hlf * 8 + r;
      O[((size_t)(bb * TT + t)) * DM + h * HD + d] = (bf16)o[j][r];
    }
  }
}

// ---------------------------------------------------------------------------
// Host launcher
// ---------------------------------------------------------------------------
extern "C" void kernel_launch(void* const* d_in, const int* in_sizes, int n_in,
                              void* d_out, int out_size, void* d_ws, size_t ws_size,
                              hipStream_t stream) {
  const float* x      = (const float*)d_in[0];
  const float* ln1_g  = (const float*)d_in[1];
  const float* ln1_b  = (const float*)d_in[2];
  const float* qkv_w  = (const float*)d_in[3];
  const float* qkv_b  = (const float*)d_in[4];
  const float* proj_w = (const float*)d_in[5];
  const float* proj_b = (const float*)d_in[6];
  const float* ln2_g  = (const float*)d_in[7];
  const float* ln2_b  = (const float*)d_in[8];
  const float* mlp_w1 = (const float*)d_in[9];
  const float* mlp_b1 = (const float*)d_in[10];
  const float* mlp_w2 = (const float*)d_in[11];
  const float* mlp_b2 = (const float*)d_in[12];

  char* p = (char*)d_ws;
  auto alloc = [&](size_t bytes) -> char* {
    char* r = p;
    p += (bytes + 255) & ~(size_t)255;
    return r;
  };
  bf16* h1   = (bf16*)alloc((size_t)BT * DM * 2);
  bf16* qb   = (bf16*)alloc((size_t)BT * DM * 2);
  bf16* kb   = (bf16*)alloc((size_t)BT * DM * 2);
  bf16* vb   = (bf16*)alloc((size_t)BT * DM * 2);   // [B,H,HD,T]
  bf16* attn = (bf16*)alloc((size_t)BT * DM * 2);
  bf16* h2   = (bf16*)alloc((size_t)BT * DM * 2);
  bf16* ff   = (bf16*)alloc((size_t)BT * DFF * 2);
  float* x1  = (float*)alloc((size_t)BT * DM * 4);
  bf16* wqkv = (bf16*)alloc((size_t)DM * 3 * DM * 2);  // transposed [3DM][DM]
  bf16* wprj = (bf16*)alloc((size_t)DM * DM * 2);      // transposed
  bf16* w1b  = (bf16*)alloc((size_t)DM * DFF * 2);     // transposed [DFF][DM]
  bf16* w2b  = (bf16*)alloc((size_t)DFF * DM * 2);     // transposed [DM][DFF]

  auto cvtT = [&](const float* s, bf16* d, int K, int N) {
    cvt_transpose_kernel<<<dim3(N / 32, K / 32), 256, 0, stream>>>(s, d, K, N);
  };
  cvtT(qkv_w,  wqkv, DM,  3 * DM);
  cvtT(proj_w, wprj, DM,  DM);
  cvtT(mlp_w1, w1b,  DM,  DFF);
  cvtT(mlp_w2, w2b,  DFF, DM);

  // ln1 -> h1 (bf16)
  layernorm_kernel<<<BT, 256, 0, stream>>>(x, ln1_g, ln1_b, h1);

  // qkv = h1 @ qkv_w + b, split to q/k [B,H,T,64], v [B,H,64,T]
  gemm_bf16_kernel<0><<<dim3((3 * DM) / 64, BT / 64), 128, 0, stream>>>(
      h1, wqkv, qkv_b, nullptr, nullptr, qb, kb, vb, BT, 3 * DM, DM);

  // flash attention -> attn [BT, DM] bf16
  attention_kernel<<<(BB * NH * (TT / 16)) / 4, 128, 0, stream>>>(qb, kb, vb, attn);

  // x1 = x + attn @ proj_w + proj_b
  gemm_bf16_kernel<1><<<dim3(DM / 64, BT / 64), 128, 0, stream>>>(
      attn, wprj, proj_b, x, x1, nullptr, nullptr, nullptr, BT, DM, DM);

  // ln2 -> h2 (bf16)
  layernorm_kernel<<<BT, 256, 0, stream>>>(x1, ln2_g, ln2_b, h2);

  // ff = gelu(h2 @ w1 + b1) (bf16)
  gemm_bf16_kernel<2><<<dim3(DFF / 64, BT / 64), 128, 0, stream>>>(
      h2, w1b, mlp_b1, nullptr, nullptr, ff, nullptr, nullptr, BT, DFF, DM);

  // out = x1 + ff @ w2 + b2 (f32)
  gemm_bf16_kernel<3><<<dim3(DM / 64, BT / 64), 128, 0, stream>>>(
      ff, w2b, mlp_b2, x1, (float*)d_out, nullptr, nullptr, nullptr, BT, DM, DFF);
}
